// QLSTM_65481071403086
// MI455X (gfx1250) — matrix operand order repaired and checked by reading, not compiled, not run on hardware
//
#include <hip/hip_runtime.h>

typedef _Float16 h16;
typedef __attribute__((ext_vector_type(16))) _Float16 v16h;
typedef __attribute__((ext_vector_type(8)))  _Float16 v8h;
typedef __attribute__((ext_vector_type(8)))  float    v8f;

#define S_LEN 512
#define BATCH 64
#define DIN   512
#define HID   512
#define NGATE 2048          // 4 gates * HID
#define KX    512           // K of the x-part GEMM
#define NWG   32            // persistent workgroups in phase 2
#define HS    16            // HID / NWG hidden units per WG
#define WSTR  520           // LDS row stride in halves (512 + 8 pad -> conflict-free)

union F16x16 { v16h v; v8h h[2]; };
union F16x8  { v8h  v; h16 e[8]; };

__device__ __forceinline__ float sigm(float x) { return 1.f / (1.f + __expf(-x)); }
__device__ __forceinline__ float tanh_fast(float x) {
  float t = __expf(-2.f * fabsf(x));
  float r = (1.f - t) / (1.f + t);
  return copysignf(r, x);
}

// ---------------- conversion kernels ----------------
__global__ void k_cvt_x(const float* __restrict__ x, h16* __restrict__ xh, int n) {
  int i = blockIdx.x * blockDim.x + threadIdx.x;
  int stride = gridDim.x * blockDim.x;
  for (; i < n; i += stride) xh[i] = (h16)x[i];
}

// Build transposed f16 weights WxT[n_g][k] = W_gate[k][n], WhT[n_g][k] = W_gate[512+k][n],
// concatenated bias, and zero the grid barrier.
__global__ void k_cvt_w(const float* __restrict__ Wf, const float* __restrict__ Wi,
                        const float* __restrict__ Wg, const float* __restrict__ Wo,
                        const float* __restrict__ bf, const float* __restrict__ bi,
                        const float* __restrict__ bg, const float* __restrict__ bo,
                        h16* __restrict__ WxT, h16* __restrict__ WhT,
                        float* __restrict__ biascat, unsigned* __restrict__ bar) {
  int idx = blockIdx.x * 256 + threadIdx.x;       // covers 2048*512
  if (idx == 0) { bar[0] = 0u; bar[1] = 0u; }
  int ng = idx >> 9, k = idx & 511;
  int gate = ng >> 9, n = ng & 511;
  const float* W = (gate == 0) ? Wf : (gate == 1) ? Wi : (gate == 2) ? Wg : Wo;
  const float* B = (gate == 0) ? bf : (gate == 1) ? bi : (gate == 2) ? bg : bo;
  WxT[idx] = (h16)W[(size_t)k * HID + n];
  WhT[idx] = (h16)W[(size_t)(512 + k) * HID + n];
  if (k == 0) biascat[ng] = B[n];
}

// ---------------- phase 1: preact = x @ Wx + b (fully parallel WMMA GEMM) ----------------
// Output layout: preact[s][n_g][b]  (f16) so phase 2 adds it with one 16B load per tile.
__global__ void __launch_bounds__(256) k_phase1(const h16* __restrict__ xh,
                                                const h16* __restrict__ WxT,
                                                const float* __restrict__ biascat,
                                                h16* __restrict__ preact) {
  const int lane = threadIdx.x & 31;
  const int wid  = blockIdx.x * 8 + (threadIdx.x >> 5);   // 65536 waves total
  const int Ngrp = wid & 31;                              // 32 N-groups of 64 cols
  const int Mt   = wid >> 5;                              // 2048 M-tiles of 16 rows
  const int row0 = Mt * 16;
  const int n0   = Ngrp * 64;
  const int m = lane & 15, c = lane >> 4;

  v8f acc[4] = {};
  const h16* arow = xh + (size_t)(row0 + m) * KX;
  for (int kc = 0; kc < KX; kc += 32) {
    F16x16 a;
    a.h[0] = *(const v8h*)(arow + kc + c * 8);
    a.h[1] = *(const v8h*)(arow + kc + 16 + c * 8);
#pragma unroll
    for (int t = 0; t < 4; ++t) {
      F16x16 b;
      const h16* brow = WxT + (size_t)(n0 + t * 16 + m) * KX + kc + c * 8;
      b.h[0] = *(const v8h*)brow;
      b.h[1] = *(const v8h*)(brow + 16);
      acc[t] = __builtin_amdgcn_wmma_f32_16x16x32_f16(false, a.v, false, b.v,
                                                      (short)0, acc[t], false, false);
    }
  }
  const int s = row0 >> 6;                 // 16 | 64, so s constant per tile
  const int bb = (row0 & 63) + c * 8;
#pragma unroll
  for (int t = 0; t < 4; ++t) {
    int n = n0 + t * 16 + m;
    float bn = biascat[n];
    F16x8 o;
#pragma unroll
    for (int r = 0; r < 8; ++r) o.e[r] = (h16)(acc[t][r] + bn);
    *(v8h*)(preact + ((size_t)s * NGATE + n) * BATCH + bb) = o.v;
  }
}

// ---------------- grid barrier (device-scope, sense via generation counter) ----------------
__device__ __forceinline__ void grid_barrier(unsigned* bar) {
  __threadfence();
  __syncthreads();
  if (threadIdx.x == 0) {
    unsigned gen = __hip_atomic_load(bar + 1, __ATOMIC_RELAXED, __HIP_MEMORY_SCOPE_AGENT);
    unsigned a = __hip_atomic_fetch_add(bar, 1u, __ATOMIC_ACQ_REL, __HIP_MEMORY_SCOPE_AGENT);
    if (a == NWG - 1) {
      __hip_atomic_store(bar, 0u, __ATOMIC_RELAXED, __HIP_MEMORY_SCOPE_AGENT);
      __hip_atomic_store(bar + 1, gen + 1u, __ATOMIC_RELEASE, __HIP_MEMORY_SCOPE_AGENT);
    } else {
      while (__hip_atomic_load(bar + 1, __ATOMIC_ACQUIRE, __HIP_MEMORY_SCOPE_AGENT) == gen) {
        __builtin_amdgcn_s_sleep(2);
      }
    }
  }
  __syncthreads();
}

// ---------------- phase 2: recurrent loop (persistent, 32 WGs x 256 threads) ----------------
// hbuf is DOUBLE-BUFFERED: step s reads hbuf[s&1], writes hbuf[(s+1)&1]. One barrier per
// step makes this race-free (all step-s reads complete before anyone enters step s+1).
__global__ void __launch_bounds__(256) k_phase2(const h16* __restrict__ preact,
                                                const h16* __restrict__ WhT,
                                                h16* __restrict__ hbuf,      // (2, B, HID) f16
                                                float* __restrict__ out,     // (S, B, HID)
                                                float* __restrict__ hx,
                                                float* __restrict__ cx,
                                                unsigned* __restrict__ bar) {
  __shared__ h16   wlds[4 * HS][WSTR];      // recurrent weight slice, resident all steps
  __shared__ float glds[4][BATCH][17];      // gate staging (padded)
  __shared__ float clds[BATCH][HS];         // cell state (persistent across steps)

  const int g   = blockIdx.x;
  const int j0  = g * HS;
  const int tid = threadIdx.x;
  const int lane = tid & 31;
  const int w = tid >> 5;
  const int Mt = w & 3;                     // batch tile (4 x 16 rows)
  const int t0 = (w >> 2) * 2;              // this wave handles gates t0, t0+1
  const int m = lane & 15, c = lane >> 4;

  // Load Wh slice into LDS: rows (gate*16 + j) of WhT, 512 halves each.
  for (int cid = tid; cid < 64 * 64; cid += 256) {   // 64 rows * 64 16B-chunks
    int rr = cid >> 6, ch = cid & 63;
    int gate = rr >> 4, j = rr & 15;
    v8h src = *(const v8h*)(WhT + (size_t)(gate * HID + j0 + j) * HID + ch * 8);
    *(v8h*)&wlds[rr][ch * 8] = src;
  }
  for (int i = tid; i < BATCH * HS; i += 256) clds[i >> 4][i & 15] = 0.f;
  __syncthreads();

  for (int s = 0; s < S_LEN; ++s) {
    const h16* hrd = hbuf + (size_t)(s & 1) * BATCH * HID;         // read buffer
    h16*       hwr = hbuf + (size_t)((s + 1) & 1) * BATCH * HID;   // write buffer
    v8f acc0 = {}, acc1 = {};
    if (s > 0) {                            // h_0 == 0 -> skip recurrent GEMM at s==0
      const h16* arow = hrd + (size_t)(Mt * 16 + m) * HID;
      for (int kc = 0; kc < HID; kc += 32) {
        F16x16 a, b0, b1;
        a.h[0]  = *(const v8h*)(arow + kc + c * 8);
        a.h[1]  = *(const v8h*)(arow + kc + 16 + c * 8);
        b0.h[0] = *(const v8h*)&wlds[t0 * 16 + m][kc + c * 8];
        b0.h[1] = *(const v8h*)&wlds[t0 * 16 + m][kc + 16 + c * 8];
        b1.h[0] = *(const v8h*)&wlds[t0 * 16 + 16 + m][kc + c * 8];
        b1.h[1] = *(const v8h*)&wlds[t0 * 16 + 16 + m][kc + 16 + c * 8];
        acc0 = __builtin_amdgcn_wmma_f32_16x16x32_f16(false, a.v, false, b0.v,
                                                      (short)0, acc0, false, false);
        acc1 = __builtin_amdgcn_wmma_f32_16x16x32_f16(false, a.v, false, b1.v,
                                                      (short)0, acc1, false, false);
      }
    }
    // Add precomputed x-part (+bias): one 16B load per tile, contiguous along b.
    const h16* p0 = preact + (((size_t)s * NGATE) + (size_t)t0 * HID + j0 + m) * BATCH + Mt * 16 + c * 8;
    const h16* p1 = preact + (((size_t)s * NGATE) + (size_t)(t0 + 1) * HID + j0 + m) * BATCH + Mt * 16 + c * 8;
    if (s + 1 < S_LEN) __builtin_prefetch(p0 + (size_t)NGATE * BATCH, 0, 1);
    F16x8 q0, q1;
    q0.v = *(const v8h*)p0;
    q1.v = *(const v8h*)p1;
#pragma unroll
    for (int r = 0; r < 8; ++r) {
      acc0[r] += (float)q0.e[r];
      acc1[r] += (float)q1.e[r];
      glds[t0][Mt * 16 + c * 8 + r][m]     = acc0[r];
      glds[t0 + 1][Mt * 16 + c * 8 + r][m] = acc1[r];
    }
    __syncthreads();

    // Elementwise LSTM cell update; write h (f16 broadcast buffer) + fp32 outputs.
    {
      int j = tid & 15, b0r = tid >> 4;
      for (int bb = b0r; bb < BATCH; bb += 16) {
        float fg = sigm(glds[0][bb][j]);
        float ig = sigm(glds[1][bb][j]);
        float gg = tanh_fast(glds[2][bb][j]);
        float og = sigm(glds[3][bb][j]);
        float cc = fg * clds[bb][j] + ig * gg;
        float hh = og * tanh_fast(cc);
        clds[bb][j] = cc;
        out[((size_t)s * BATCH + bb) * HID + j0 + j] = hh;
        hwr[bb * HID + j0 + j] = (h16)hh;
        if (s == S_LEN - 1) {
          hx[bb * HID + j0 + j] = hh;
          cx[bb * HID + j0 + j] = cc;
        }
      }
    }
    grid_barrier(bar);   // h fully published before any WG starts step s+1
  }
}

// ---------------- launch ----------------
extern "C" void kernel_launch(void* const* d_in, const int* in_sizes, int n_in,
                              void* d_out, int out_size, void* d_ws, size_t ws_size,
                              hipStream_t stream) {
  (void)in_sizes; (void)n_in; (void)out_size; (void)ws_size;
  const float* x  = (const float*)d_in[0];
  const float* Wf = (const float*)d_in[1];
  const float* bf = (const float*)d_in[2];
  const float* Wi = (const float*)d_in[3];
  const float* bi = (const float*)d_in[4];
  const float* Wg = (const float*)d_in[5];
  const float* bg = (const float*)d_in[6];
  const float* Wo = (const float*)d_in[7];
  const float* bo = (const float*)d_in[8];
  float* out = (float*)d_out;
  char* ws = (char*)d_ws;

  unsigned* bar  = (unsigned*)ws;
  size_t off = 1024;
  h16*   xh      = (h16*)(ws + off);            off += (size_t)S_LEN * BATCH * DIN * 2;   // 32 MB
  h16*   WxT     = (h16*)(ws + off);            off += (size_t)NGATE * KX * 2;            // 2 MB
  h16*   WhT     = (h16*)(ws + off);            off += (size_t)NGATE * HID * 2;           // 2 MB
  float* biascat = (float*)(ws + off);          off += (size_t)NGATE * 4;                 // 8 KB
  h16*   hbuf    = (h16*)(ws + off);            off += (size_t)2 * BATCH * HID * 2;       // 128 KB (double-buffered)
  h16*   preact  = (h16*)(ws + off);            // 128 MB: (S, NGATE, B) f16

  float* hx = out + (size_t)S_LEN * BATCH * HID;
  float* cx = hx + (size_t)BATCH * HID;

  k_cvt_x<<<8192, 256, 0, stream>>>(x, xh, S_LEN * BATCH * DIN);
  k_cvt_w<<<4096, 256, 0, stream>>>(Wf, Wi, Wg, Wo, bf, bi, bg, bo, WxT, WhT, biascat, bar);
  k_phase1<<<8192, 256, 0, stream>>>(xh, WxT, biascat, preact);
  k_phase2<<<NWG, 256, 0, stream>>>(preact, WhT, hbuf, out, hx, cx, bar);
}
